// MultiHeadAttention_47055661695107
// MI455X (gfx1250) — matrix operand order repaired and checked by reading, not compiled, not run on hardware
//
#include <hip/hip_runtime.h>
#include <stdint.h>

// Problem constants (match reference)
#define BB 2
#define SS 2048
#define EE 1024
#define HH 16
#define DD 64
#define MM (BB * SS)   // 4096

typedef __attribute__((ext_vector_type(16))) _Float16 v16h;
typedef __attribute__((ext_vector_type(8)))  _Float16 v8h;
typedef __attribute__((ext_vector_type(8)))  float    v8f;
typedef _Float16 h16;

// ---------------------------------------------------------------------------
// WMMA wrapper: D = A(16x32 f16) * B(32x16 f16) + C(16x16 f32)
// ---------------------------------------------------------------------------
__device__ __forceinline__ v8f wmma_f16(v16h a, v16h b, v8f c) {
  return __builtin_amdgcn_wmma_f32_16x16x32_f16(
      /*neg_a=*/false, a, /*neg_b=*/false, b,
      /*c_mod=*/(short)0, c, /*reuse_a=*/false, /*reuse_b=*/false);
}

// ---------------------------------------------------------------------------
// Fragment loaders (ISA 7.12.2 16-bit layouts).
// A-frag: lane l holds row l%16; K chunks [k+8*hs,+8) and [k+16+8*hs,+8).
// B-frag: lane l holds col l%16; K chunks [k+8*hs,+8) and [k+16+8*hs,+8).
// ---------------------------------------------------------------------------
__device__ __forceinline__ v16h ldA_f16(const h16* __restrict__ p, int hs) {
  p += hs * 8;
  v8h lo = *(const v8h*)(p);       // 16B
  v8h hi = *(const v8h*)(p + 16);  // 16B
  v16h r;
#pragma unroll
  for (int i = 0; i < 8; ++i) { r[i] = lo[i]; r[8 + i] = hi[i]; }
  return r;
}

__device__ __forceinline__ v16h ldB_f16(const h16* __restrict__ p, int hs) {
  return ldA_f16(p, hs);
}

__device__ __forceinline__ v16h cvt16(const float* __restrict__ p) {
  v16h r;
#pragma unroll
  for (int i = 0; i < 16; i += 4) {
    float4 f = *(const float4*)(p + i);
    r[i]     = (_Float16)f.x; r[i + 1] = (_Float16)f.y;
    r[i + 2] = (_Float16)f.z; r[i + 3] = (_Float16)f.w;
  }
  return r;
}

__device__ __forceinline__ v16h ldA_f32(const float* __restrict__ p, int hs) {
  p += hs * 8;
  v16h r;
#pragma unroll
  for (int i = 0; i < 8; i += 4) {
    float4 f = *(const float4*)(p + i);
    r[i]     = (_Float16)f.x; r[i + 1] = (_Float16)f.y;
    r[i + 2] = (_Float16)f.z; r[i + 3] = (_Float16)f.w;
    float4 g = *(const float4*)(p + 16 + i);
    r[8 + i]  = (_Float16)g.x; r[9 + i]  = (_Float16)g.y;
    r[10 + i] = (_Float16)g.z; r[11 + i] = (_Float16)g.w;
  }
  return r;
}

__device__ __forceinline__ v16h ldB_f32(const float* __restrict__ p, int hs) {
  return ldA_f32(p, hs);
}

// ---------------------------------------------------------------------------
// Kernel 1: y = x @ W^T + bias, output f16.
// mode 0 -> out[b][h][s][d]  (Q, K)       mode 1 -> out[b][h][d][s]  (V^T)
// ---------------------------------------------------------------------------
__device__ __forceinline__ void store_proj(v8f c, int rowBase, int col,
                                           const float* __restrict__ bias,
                                           h16* __restrict__ out, int mode,
                                           int hs) {
  float bv = bias[col];
#pragma unroll
  for (int r = 0; r < 8; ++r) {
    int row = rowBase + r + hs * 8;          // row = b*SS + s
    float val = c[r] + bv;
    int bb = row >> 11, s = row & (SS - 1);
    int hh = col >> 6,  d = col & (DD - 1);
    size_t off = (mode == 0)
        ? (((size_t)(bb * HH + hh) * SS + s) * DD + d)
        : (((size_t)(bb * HH + hh) * DD + d) * SS + s);
    out[off] = (_Float16)val;
  }
}

__global__ void __launch_bounds__(256)
proj_gemm(const float* __restrict__ X, const float* __restrict__ W,
          const float* __restrict__ bias, h16* __restrict__ out, int mode) {
  const int tid  = threadIdx.x;
  const int wave = tid >> 5;
  const int lane = tid & 31;
  const int hs   = lane >> 4;
  const int l16  = lane & 15;

  const int tM = blockIdx.x & 31;
  const int tN = blockIdx.x >> 5;
  const int wm = wave & 3, wn = wave >> 2;
  const int rowBase = tM * 128 + wm * 32;
  const int colBase = tN * 64 + wn * 32;

  const float* Ar0 = X + (size_t)(rowBase + l16) * EE;
  const float* Ar1 = Ar0 + (size_t)16 * EE;
  const float* Wr0 = W + (size_t)(colBase + l16) * EE;
  const float* Wr1 = Wr0 + (size_t)16 * EE;

  v8f c00 = {}, c01 = {}, c10 = {}, c11 = {};

  for (int k = 0; k < EE; k += 32) {
    __builtin_prefetch(Ar0 + k + 64, 0, 0);
    __builtin_prefetch(Wr0 + k + 64, 0, 0);
    v16h a0 = ldA_f32(Ar0 + k, hs);
    v16h a1 = ldA_f32(Ar1 + k, hs);
    v16h b0 = ldB_f32(Wr0 + k, hs);
    v16h b1 = ldB_f32(Wr1 + k, hs);
    c00 = wmma_f16(a0, b0, c00);
    c01 = wmma_f16(a0, b1, c01);
    c10 = wmma_f16(a1, b0, c10);
    c11 = wmma_f16(a1, b1, c11);
  }

  store_proj(c00, rowBase,      colBase + l16,      bias, out, mode, hs);
  store_proj(c01, rowBase,      colBase + 16 + l16, bias, out, mode, hs);
  store_proj(c10, rowBase + 16, colBase + l16,      bias, out, mode, hs);
  store_proj(c11, rowBase + 16, colBase + 16 + l16, bias, out, mode, hs);
}

// ---------------------------------------------------------------------------
// Kernel 2: streaming-softmax attention with async double-buffered K/V tiles.
// Block = 8 waves sharing one (b,h) and a 128-query slab; wave = 16-row Q
// tile. K tile [32k x 64d] and V tile [64d x 32k] staged into LDS with
// GLOBAL_LOAD_ASYNC_TO_LDS_B128 (ASYNCcnt), double buffered.
// Grid: B*H*(S/128) = 512 blocks of 256.
// ---------------------------------------------------------------------------
__global__ void __launch_bounds__(256)
attn_kernel(const h16* __restrict__ Qh, const h16* __restrict__ Kh,
            const h16* __restrict__ Vt, h16* __restrict__ Ah) {
  const int tid  = threadIdx.x;
  const int wave = tid >> 5;
  const int lane = tid & 31;
  const int hs   = lane >> 4;
  const int l16  = lane & 15;

  const int idx = blockIdx.x;            // 0..511
  const int b   = idx >> 8;
  const int h   = (idx >> 4) & 15;
  const int qb  = idx & 15;
  const int qbase = qb * 128 + wave * 16;

  const h16* Qb = Qh + (size_t)(b * HH + h) * SS * DD;
  const h16* Kb = Kh + (size_t)(b * HH + h) * SS * DD;
  const h16* Vb = Vt + (size_t)(b * HH + h) * DD * SS;

  __shared__ h16 Kt [2][32][DD];   // key-major: rows contiguous in d
  __shared__ h16 Vtl[2][DD][32];   // d-major: rows contiguous in keys
  __shared__ h16 Pst[8][16 * 32];
  h16* P = Pst[wave];

  // per-thread 16B async-copy slices
  const int krow = tid >> 3, kcol = (tid & 7) * 8;  // K tile 32x64
  const int vrow = tid >> 2, vcol = (tid & 3) * 8;  // V tile 64x32

  auto issue_tiles = [&](int buf, int j) {
    unsigned lk = (unsigned)(uintptr_t)&Kt[buf][krow][kcol];
    unsigned long long gk =
        (unsigned long long)(uintptr_t)(Kb + (size_t)(j + krow) * DD + kcol);
    asm volatile("global_load_async_to_lds_b128 %0, %1, off"
                 :: "v"(lk), "v"(gk) : "memory");
    unsigned lv = (unsigned)(uintptr_t)&Vtl[buf][vrow][vcol];
    unsigned long long gv =
        (unsigned long long)(uintptr_t)(Vb + (size_t)vrow * SS + j + vcol);
    asm volatile("global_load_async_to_lds_b128 %0, %1, off"
                 :: "v"(lv), "v"(gv) : "memory");
  };

  // Q fragments (16x64 over two K=32 frags), loaded once from global
  const h16* qrow = Qb + (size_t)(qbase + l16) * DD;
  const v16h qf0 = ldA_f16(qrow, hs);
  const v16h qf1 = ldA_f16(qrow + 32, hs);

  v8f o0 = {}, o1 = {}, o2 = {}, o3 = {};
  float mr[8], lr[8];
#pragma unroll
  for (int r = 0; r < 8; ++r) { mr[r] = -1e30f; lr[r] = 0.f; }

  issue_tiles(0, 0);

  for (int j = 0; j < SS; j += 32) {
    const int cur = (j >> 5) & 1;
    asm volatile("s_wait_asynccnt 0x0" ::: "memory");
    __syncthreads();
    if (j + 32 < SS) issue_tiles(cur ^ 1, j + 32);

    v8f s0 = {}, s1 = {};
    s0 = wmma_f16(qf0, ldB_f16(&Kt[cur][l16][0],       hs), s0);
    s0 = wmma_f16(qf1, ldB_f16(&Kt[cur][l16][32],      hs), s0);
    s1 = wmma_f16(qf0, ldB_f16(&Kt[cur][16 + l16][0],  hs), s1);
    s1 = wmma_f16(qf1, ldB_f16(&Kt[cur][16 + l16][32], hs), s1);

#pragma unroll
    for (int r = 0; r < 8; ++r) {
      float a = s0[r] * 0.125f;   // 1/sqrt(64)
      float c = s1[r] * 0.125f;
      float v = fmaxf(a, c);
      v = fmaxf(v, __shfl_xor(v, 1));
      v = fmaxf(v, __shfl_xor(v, 2));
      v = fmaxf(v, __shfl_xor(v, 4));
      v = fmaxf(v, __shfl_xor(v, 8));
      float mn   = fmaxf(mr[r], v);
      float corr = __expf(mr[r] - mn);
      float p0   = __expf(a - mn);
      float p1   = __expf(c - mn);
      float sv = p0 + p1;
      sv += __shfl_xor(sv, 1);
      sv += __shfl_xor(sv, 2);
      sv += __shfl_xor(sv, 4);
      sv += __shfl_xor(sv, 8);
      lr[r] = lr[r] * corr + sv;
      mr[r] = mn;
      o0[r] *= corr; o1[r] *= corr; o2[r] *= corr; o3[r] *= corr;
      int row = r + hs * 8;
      P[row * 32 + l16]      = (_Float16)p0;
      P[row * 32 + 16 + l16] = (_Float16)p1;
    }

    v16h pa = ldA_f16(P + l16 * 32, hs);

    o0 = wmma_f16(pa, ldB_f16(&Vtl[cur][l16][0],      hs), o0);
    o1 = wmma_f16(pa, ldB_f16(&Vtl[cur][16 + l16][0], hs), o1);
    o2 = wmma_f16(pa, ldB_f16(&Vtl[cur][32 + l16][0], hs), o2);
    o3 = wmma_f16(pa, ldB_f16(&Vtl[cur][48 + l16][0], hs), o3);
  }

#pragma unroll
  for (int r = 0; r < 8; ++r) {
    float inv = 1.0f / lr[r];
    int row = qbase + r + hs * 8;
    size_t base = ((size_t)b * SS + row) * EE + h * DD;
    Ah[base + l16]      = (_Float16)(o0[r] * inv);
    Ah[base + 16 + l16] = (_Float16)(o1[r] * inv);
    Ah[base + 32 + l16] = (_Float16)(o2[r] * inv);
    Ah[base + 48 + l16] = (_Float16)(o3[r] * inv);
  }
}

// ---------------------------------------------------------------------------
// Kernel 3: out = Ah @ Wo^T + bo, f32 output (row-major [B*S, E]).
// ---------------------------------------------------------------------------
__global__ void __launch_bounds__(256)
oproj_gemm(const h16* __restrict__ Ah, const float* __restrict__ Wo,
           const float* __restrict__ bo, float* __restrict__ out) {
  const int tid  = threadIdx.x;
  const int wave = tid >> 5;
  const int lane = tid & 31;
  const int hs   = lane >> 4;
  const int l16  = lane & 15;

  const int tM = blockIdx.x & 31;
  const int tN = blockIdx.x >> 5;
  const int wm = wave & 3, wn = wave >> 2;
  const int rowBase = tM * 128 + wm * 32;
  const int colBase = tN * 64 + wn * 32;

  const h16*   Ar0 = Ah + (size_t)(rowBase + l16) * EE;
  const h16*   Ar1 = Ar0 + (size_t)16 * EE;
  const float* Wr0 = Wo + (size_t)(colBase + l16) * EE;
  const float* Wr1 = Wr0 + (size_t)16 * EE;

  v8f c00 = {}, c01 = {}, c10 = {}, c11 = {};

  for (int k = 0; k < EE; k += 32) {
    __builtin_prefetch(Wr0 + k + 64, 0, 0);
    v16h a0 = ldA_f16(Ar0 + k, hs);
    v16h a1 = ldA_f16(Ar1 + k, hs);
    v16h b0 = ldB_f32(Wr0 + k, hs);
    v16h b1 = ldB_f32(Wr1 + k, hs);
    c00 = wmma_f16(a0, b0, c00);
    c01 = wmma_f16(a0, b1, c01);
    c10 = wmma_f16(a1, b0, c10);
    c11 = wmma_f16(a1, b1, c11);
  }

  const float bv0 = bo[colBase + l16];
  const float bv1 = bo[colBase + 16 + l16];
#pragma unroll
  for (int r = 0; r < 8; ++r) {
    int row0 = rowBase + r + hs * 8;
    int row1 = row0 + 16;
    out[(size_t)row0 * EE + colBase + l16]      = c00[r] + bv0;
    out[(size_t)row0 * EE + colBase + 16 + l16] = c01[r] + bv1;
    out[(size_t)row1 * EE + colBase + l16]      = c10[r] + bv0;
    out[(size_t)row1 * EE + colBase + 16 + l16] = c11[r] + bv1;
  }
}

// ---------------------------------------------------------------------------
// Host launcher
// ---------------------------------------------------------------------------
extern "C" void kernel_launch(void* const* d_in, const int* in_sizes, int n_in,
                              void* d_out, int out_size, void* d_ws,
                              size_t ws_size, hipStream_t stream) {
  const float* x  = (const float*)d_in[0];
  const float* Wq = (const float*)d_in[1];
  const float* bq = (const float*)d_in[2];
  const float* Wk = (const float*)d_in[3];
  const float* bk = (const float*)d_in[4];
  const float* Wv = (const float*)d_in[5];
  const float* bv = (const float*)d_in[6];
  const float* Wo = (const float*)d_in[7];
  const float* bo = (const float*)d_in[8];

  const size_t elems = (size_t)BB * HH * SS * DD;  // 4M f16 elements each
  h16* Qh = (h16*)d_ws;
  h16* Kh = Qh + elems;
  h16* Vt = Kh + elems;
  h16* Ah = Vt + elems;

  dim3 blk(256);
  dim3 grd(512);

  proj_gemm<<<grd, blk, 0, stream>>>(x, Wq, bq, Qh, 0);
  proj_gemm<<<grd, blk, 0, stream>>>(x, Wk, bk, Kh, 0);
  proj_gemm<<<grd, blk, 0, stream>>>(x, Wv, bv, Vt, 1);  // V transposed
  attn_kernel<<<grd, blk, 0, stream>>>(Qh, Kh, Vt, Ah);
  oproj_gemm<<<grd, blk, 0, stream>>>(Ah, Wo, bo, (float*)d_out);
}
